// AdditiveAttention_22058952032611
// MI455X (gfx1250) — matrix operand order repaired and checked by reading, not compiled, not run on hardware
//
#include <hip/hip_runtime.h>
#include <hip/hip_bf16.h>

typedef __attribute__((ext_vector_type(16))) __bf16 v16bf;
typedef __attribute__((ext_vector_type(8)))  __bf16 v8bf;
typedef __attribute__((ext_vector_type(4)))  __bf16 v4bf;
typedef __attribute__((ext_vector_type(8)))  float  v8f;

#define BB   4
#define QQ   256
#define KVN  1024
#define HD   128
#define VSD  256

// ---- gfx1250 feature probes ------------------------------------------------
#if __has_builtin(__builtin_amdgcn_global_load_async_to_lds_b128) && \
    __has_builtin(__builtin_amdgcn_s_wait_asynccnt)
#define USE_ASYNC_LDS 1
#endif

// Builtin signature (probe-derived from hipcc diagnostic):
//   void __builtin_amdgcn_global_load_async_to_lds_b128(
//       int4 AS1* src, int4 AS3* dst, imm int offset, imm int cpol)
typedef int v4i __attribute__((vector_size(16)));
typedef __attribute__((address_space(1))) v4i v4i_g;
typedef __attribute__((address_space(3))) v4i v4i_l;

__device__ __forceinline__ v4i_g* to_global_v4(const void* p) {
  return (v4i_g*)(uint64_t)p;
}
__device__ __forceinline__ v4i_l* to_local_v4(void* p) {
  // generic LDS pointer: low 32 bits are the LDS offset
  return (v4i_l*)(uint32_t)(uint64_t)p;
}

__device__ __forceinline__ float fast_tanh(float x) {
#if __has_builtin(__builtin_amdgcn_tanhf)
  return __builtin_amdgcn_tanhf(x);     // gfx1250 v_tanh_f32 (TRANS)
#else
  return tanhf(x);
#endif
}

// ---------------------------------------------------------------------------
// f32 -> bf16 straight conversion (vectorized, n multiple of 4)
// ---------------------------------------------------------------------------
__global__ __launch_bounds__(256)
void cvt_bf16(const float* __restrict__ in, __bf16* __restrict__ out, int n) {
  const int i = (blockIdx.x * blockDim.x + threadIdx.x) * 4;
  if (i < n) {
    const float4 v = *(const float4*)(in + i);
    v4bf o;
    o[0] = (__bf16)v.x; o[1] = (__bf16)v.y; o[2] = (__bf16)v.z; o[3] = (__bf16)v.w;
    *(v4bf*)(out + i) = o;
  }
}

// ---------------------------------------------------------------------------
// f32 [R][C] -> bf16 [C][R] transpose-convert via 32x32 LDS tile (batched)
// ---------------------------------------------------------------------------
__global__ __launch_bounds__(256)
void cvt_transpose_bf16(const float* __restrict__ in, __bf16* __restrict__ out,
                        int R, int C, long sIn, long sOut) {
  const int batch = blockIdx.z;
  in  += (size_t)batch * sIn;
  out += (size_t)batch * sOut;
  __shared__ float tile[32][33];
  const int c0 = blockIdx.x * 32, r0 = blockIdx.y * 32;
  const int tx = threadIdx.x & 31, ty = threadIdx.x >> 5;   // 8 rows of 32
#pragma unroll
  for (int rr = ty; rr < 32; rr += 8)
    tile[rr][tx] = in[(size_t)(r0 + rr) * C + c0 + tx];
  __syncthreads();
#pragma unroll
  for (int cc = ty; cc < 32; cc += 8)
    out[(size_t)(c0 + cc) * R + r0 + tx] = (__bf16)tile[tx][cc];
}

// ---------------------------------------------------------------------------
// C(f32, MxN) = A(bf16, MxK rowmaj) * BT(bf16, NxK rowmaj)^T, one wave per
// 16x16 tile, v_wmma_f32_16x16x32_bf16, K unrolled by 64 (2 WMMA / iter).
// Fragment layouts used here (A/B: contiguous k, not ISA table order) (half = lane>>4, r = lane&15):
//   A : lane r = row, bf16 K chunk {half*16..+15} -> contiguous 32B from A row
//   B : lane r = col, K chunk {half*16..+15} -> contiguous 32B from BT row
//   C : VGPR v -> row half*8+v, col r
// ---------------------------------------------------------------------------
__device__ __forceinline__ v16bf load_a_frag(const __bf16* A, int K,
                                             int row, int k0, int half) {
  const __bf16* p = A + (size_t)row * K + k0 + half * 16;
  return *(const v16bf*)p;
}
__device__ __forceinline__ v16bf load_b_frag(const __bf16* BT, int K,
                                             int col, int k0, int half) {
  return *(const v16bf*)(BT + (size_t)col * K + k0 + half * 16);
}

__global__ __launch_bounds__(256)
void wmma_gemm_bf16(const __bf16* __restrict__ A,
                    const __bf16* __restrict__ BT,
                    float* __restrict__ C,
                    int M, int N, int K,
                    long strideA, long strideBT, long strideC) {
  const int batch = blockIdx.y;
  A  += (size_t)batch * strideA;
  BT += (size_t)batch * strideBT;
  C  += (size_t)batch * strideC;

  const int lane = threadIdx.x & 31;
  const int wave = blockIdx.x * (blockDim.x >> 5) + (threadIdx.x >> 5);
  const int tilesN = N >> 4;
  const int m0 = (wave / tilesN) << 4;
  const int n0 = (wave % tilesN) << 4;
  if (m0 >= M) return;          // whole-wave uniform exit; EXEC all-1 at WMMA

  const int half = lane >> 4;
  const int r    = lane & 15;
  const int arow = m0 + r;
  const int bcol = n0 + r;

  v8f acc = {};
  for (int k0 = 0; k0 < K; k0 += 64) {
    __builtin_prefetch(A + (size_t)arow * K + k0 + 64, 0, 1);  // global_prefetch_b8
    __builtin_prefetch(BT + (size_t)bcol * K + k0 + 64, 0, 1);
    const v16bf a0 = load_a_frag(A, K, arow, k0, half);
    const v16bf b0 = load_b_frag(BT, K, bcol, k0, half);
    const v16bf a1 = load_a_frag(A, K, arow, k0 + 32, half);
    const v16bf b1 = load_b_frag(BT, K, bcol, k0 + 32, half);
    acc = __builtin_amdgcn_wmma_f32_16x16x32_bf16(false, a0, false, b0,
                                                  (short)0, acc, false, false);
    acc = __builtin_amdgcn_wmma_f32_16x16x32_bf16(false, a1, false, b1,
                                                  (short)0, acc, false, false);
  }

#pragma unroll
  for (int v = 0; v < 8; ++v)
    C[(size_t)(m0 + half * 8 + v) * N + n0 + r] = acc[v];
}

// ---------------------------------------------------------------------------
// scores[b,q,kv] = sum_h tanh(qp[b,q,h] + kp[b,kv,h]) * wv[h], masked -1e6.
// 8 q-rows x 64 kv tile; k tile staged via GLOBAL_LOAD_ASYNC_TO_LDS_B128
// (ASYNCcnt path) when available. Inner loop: float4 ds_load + v_tanh + fma.
// ---------------------------------------------------------------------------
__global__ __launch_bounds__(256)
void scores_tanh_kernel(const float* __restrict__ qp,
                        const float* __restrict__ kp,
                        const float* __restrict__ wv,
                        const int*   __restrict__ valid,
                        float* __restrict__ scores) {
  __shared__ __align__(16) float sk[64][HD + 4];   // row = 528B = 33*16
  __shared__ __align__(16) float sq[8][HD];
  __shared__ __align__(16) float swv[HD];

  const int b  = blockIdx.x;
  const int qt = blockIdx.y;
  const int kt = blockIdx.z;

  const float* kbase = kp + ((size_t)b * KVN + (size_t)kt * 64) * HD;
#ifdef USE_ASYNC_LDS
  for (int i = threadIdx.x * 4; i < 64 * HD; i += 256 * 4) {
    const int kvv = i >> 7, h = i & (HD - 1);
    __builtin_amdgcn_global_load_async_to_lds_b128(
        to_global_v4(kbase + i), to_local_v4(&sk[kvv][h]), 0, 0);
  }
#else
  for (int i = threadIdx.x * 4; i < 64 * HD; i += 256 * 4) {
    const int kvv = i >> 7, h = i & (HD - 1);
    *(float4*)&sk[kvv][h] = *(const float4*)(kbase + i);
  }
#endif

  const float* qbase = qp + ((size_t)b * QQ + (size_t)qt * 8) * HD;
  for (int i = threadIdx.x; i < 8 * HD; i += 256)
    sq[i >> 7][i & (HD - 1)] = qbase[i];
  if (threadIdx.x < HD) swv[threadIdx.x] = wv[threadIdx.x];

#ifdef USE_ASYNC_LDS
  __builtin_amdgcn_s_wait_asynccnt(0);
#endif
  __syncthreads();

  const int kv   = threadIdx.x & 63;
  const int qs   = threadIdx.x >> 6;        // 0..3
  const int kvg  = kt * 64 + kv;
  const int vlen = valid[b];
  const bool live = kvg < vlen;

#pragma unroll
  for (int qq = 0; qq < 8; qq += 4) {
    const int q = qq + qs;
    float acc = 0.f;
#pragma unroll 4
    for (int h = 0; h < HD; h += 4) {
      const float4 k4 = *(const float4*)&sk[kv][h];
      const float4 q4 = *(const float4*)&sq[q][h];
      const float4 w4 = *(const float4*)&swv[h];
      acc += fast_tanh(q4.x + k4.x) * w4.x;
      acc += fast_tanh(q4.y + k4.y) * w4.y;
      acc += fast_tanh(q4.z + k4.z) * w4.z;
      acc += fast_tanh(q4.w + k4.w) * w4.w;
    }
    scores[((size_t)b * QQ + (size_t)qt * 8 + q) * KVN + kvg] = live ? acc : -1e6f;
  }
}

// ---------------------------------------------------------------------------
// Masked softmax over KV (in f32), emitting bf16 attention weights directly.
// One 256-thread block per (b,q) row; wave32 shfl_xor + LDS combine.
// ---------------------------------------------------------------------------
__global__ __launch_bounds__(256)
void softmax_bf16_kernel(const float* __restrict__ scores,
                         __bf16* __restrict__ attn) {
  const float* row = scores + (size_t)blockIdx.x * KVN;
  __bf16* orow = attn + (size_t)blockIdx.x * KVN;
  __shared__ float redm[8];
  __shared__ float reds[8];

  const int t = threadIdx.x, lane = t & 31, w = t >> 5;
  float v[4];
  float m = -3.4e38f;
#pragma unroll
  for (int i = 0; i < 4; ++i) { v[i] = row[t + i * 256]; m = fmaxf(m, v[i]); }
#pragma unroll
  for (int off = 16; off > 0; off >>= 1) m = fmaxf(m, __shfl_xor(m, off, 32));
  if (lane == 0) redm[w] = m;
  __syncthreads();
#pragma unroll
  for (int i = 0; i < 8; ++i) m = fmaxf(m, redm[i]);

  float s = 0.f;
#pragma unroll
  for (int i = 0; i < 4; ++i) { v[i] = __expf(v[i] - m); s += v[i]; }
#pragma unroll
  for (int off = 16; off > 0; off >>= 1) s += __shfl_xor(s, off, 32);
  if (lane == 0) reds[w] = s;
  __syncthreads();
  s = 0.f;
#pragma unroll
  for (int i = 0; i < 8; ++i) s += reds[i];

  const float inv = 1.0f / s;
#pragma unroll
  for (int i = 0; i < 4; ++i) orow[t + i * 256] = (__bf16)(v[i] * inv);
}

// ---------------------------------------------------------------------------
extern "C" void kernel_launch(void* const* d_in, const int* in_sizes, int n_in,
                              void* d_out, int out_size, void* d_ws, size_t ws_size,
                              hipStream_t stream) {
  const float* queries = (const float*)d_in[0];   // [B,Q,128]
  const float* keys    = (const float*)d_in[1];   // [B,KV,128]
  const float* values  = (const float*)d_in[2];   // [B,KV,256]
  const int*   vlens   = (const int*)  d_in[3];   // [B]
  const float* Wq      = (const float*)d_in[4];   // [128,128]
  const float* Wk      = (const float*)d_in[5];   // [128,128]
  const float* wv      = (const float*)d_in[6];   // [128]
  float* out = (float*)d_out;                     // [B,Q,256]

  // --- workspace carve-up (all sizes multiples of 16B) ---------------------
  char* w = (char*)d_ws;
  float* qp      = (float*)w;  w += (size_t)BB * QQ  * HD  * 4;   // 512 KB
  float* kp      = (float*)w;  w += (size_t)BB * KVN * HD  * 4;   // 2 MB
  float* sc      = (float*)w;  w += (size_t)BB * QQ  * KVN * 4;   // 4 MB
  __bf16* q_bf   = (__bf16*)w; w += (size_t)BB * QQ  * HD  * 2;
  __bf16* k_bf   = (__bf16*)w; w += (size_t)BB * KVN * HD  * 2;
  __bf16* WqT    = (__bf16*)w; w += (size_t)HD * HD * 2;
  __bf16* WkT    = (__bf16*)w; w += (size_t)HD * HD * 2;
  __bf16* attnbf = (__bf16*)w; w += (size_t)BB * QQ  * KVN * 2;
  __bf16* vT     = (__bf16*)w; w += (size_t)BB * VSD * KVN * 2;

  // --- operand conversion (bandwidth-trivial) ------------------------------
  cvt_bf16<<<(BB * QQ * HD) / 1024, 256, 0, stream>>>(queries, q_bf, BB * QQ * HD);
  cvt_bf16<<<(BB * KVN * HD) / 1024, 256, 0, stream>>>(keys, k_bf, BB * KVN * HD);
  cvt_transpose_bf16<<<dim3(HD / 32, HD / 32, 1), 256, 0, stream>>>(
      Wq, WqT, HD, HD, 0, 0);
  cvt_transpose_bf16<<<dim3(HD / 32, HD / 32, 1), 256, 0, stream>>>(
      Wk, WkT, HD, HD, 0, 0);
  cvt_transpose_bf16<<<dim3(VSD / 32, KVN / 32, BB), 256, 0, stream>>>(
      values, vT, KVN, VSD, (long)KVN * VSD, (long)VSD * KVN);

  // --- projections via WMMA ------------------------------------------------
  // q: (B*Q=1024) x 128 x 128 -> 512 tile-waves -> 64 blocks
  wmma_gemm_bf16<<<dim3(64, 1), 256, 0, stream>>>(
      q_bf, WqT, qp, BB * QQ, HD, HD, 0, 0, 0);
  // k: (B*KV=4096) x 128 x 128 -> 2048 tile-waves -> 256 blocks
  wmma_gemm_bf16<<<dim3(256, 1), 256, 0, stream>>>(
      k_bf, WkT, kp, BB * KVN, HD, HD, 0, 0, 0);

  // --- additive scores (dominant: 134M v_tanh, fp32) -----------------------
  scores_tanh_kernel<<<dim3(BB, QQ / 8, KVN / 64), 256, 0, stream>>>(
      qp, kp, wv, vlens, sc);

  // --- masked softmax -> bf16 attn -----------------------------------------
  softmax_bf16_kernel<<<BB * QQ, 256, 0, stream>>>(sc, attnbf);

  // --- out = attn @ values : batched 256 x 1024 x 256 ----------------------
  wmma_gemm_bf16<<<dim3(32, BB), 256, 0, stream>>>(
      attnbf, vT, out, QQ, VSD, KVN,
      (long)QQ * KVN, (long)VSD * KVN, (long)QQ * VSD);
}
